// Self_Attn_53077205844763
// MI455X (gfx1250) — matrix operand order repaired and checked by reading, not compiled, not run on hardware
//
#include <hip/hip_runtime.h>
#include <hip/hip_bf16.h>
#include <stdint.h>
#include <stddef.h>

// ---- CDNA5 WMMA vector types (gfx1250, wave32) ------------------------------
typedef __attribute__((ext_vector_type(8)))  __bf16 v8bf;
typedef __attribute__((ext_vector_type(16))) __bf16 v16bf;
typedef __attribute__((ext_vector_type(8)))  float  v8f;

#define B_  32
#define C_  1024
#define N_  512

union FragA { v16bf v; v8bf h[2]; };

__device__ __forceinline__ v8f wmma_bf16(const FragA& a, const FragA& b, v8f c) {
    return __builtin_amdgcn_wmma_f32_16x16x32_bf16(false, a.v, false, b.v,
                                                   (short)0, c, false, false);
}

// Wave-level 32x64 GEMM strip: 2 A-fragments x 4 B-fragments = 8 WMMAs / k-step.
// A rows are row-major over the contraction dim (stride = klen elements must be
// baked into the pointers by the caller); B rows likewise.
__device__ __forceinline__ void wave_gemm_32x64(
    const __bf16* __restrict__ a0, const __bf16* __restrict__ a1,
    const __bf16* __restrict__ b0, const __bf16* __restrict__ b1,
    const __bf16* __restrict__ b2, const __bf16* __restrict__ b3,
    int klen, int lane, v8f acc[2][4])
{
    const int ka = (lane >> 4) * 8;    // A layout: chunks {ka..+7} and {16+ka..+7}
    const int kb = (lane >> 4) * 8;    // B layout: chunks {kb..+7} and {16+kb..+7}
    for (int kk = 0; kk < klen; kk += 32) {
        FragA A0, A1, B0, B1, B2, B3;
        A0.h[0] = *(const v8bf*)(a0 + kk + ka);
        A0.h[1] = *(const v8bf*)(a0 + kk + 16 + ka);
        A1.h[0] = *(const v8bf*)(a1 + kk + ka);
        A1.h[1] = *(const v8bf*)(a1 + kk + 16 + ka);
        B0.h[0] = *(const v8bf*)(b0 + kk + kb);
        B0.h[1] = *(const v8bf*)(b0 + kk + 16 + kb);
        B1.h[0] = *(const v8bf*)(b1 + kk + kb);
        B1.h[1] = *(const v8bf*)(b1 + kk + 16 + kb);
        B2.h[0] = *(const v8bf*)(b2 + kk + kb);
        B2.h[1] = *(const v8bf*)(b2 + kk + 16 + kb);
        B3.h[0] = *(const v8bf*)(b3 + kk + kb);
        B3.h[1] = *(const v8bf*)(b3 + kk + 16 + kb);
        acc[0][0] = wmma_bf16(A0, B0, acc[0][0]);
        acc[0][1] = wmma_bf16(A0, B1, acc[0][1]);
        acc[0][2] = wmma_bf16(A0, B2, acc[0][2]);
        acc[0][3] = wmma_bf16(A0, B3, acc[0][3]);
        acc[1][0] = wmma_bf16(A1, B0, acc[1][0]);
        acc[1][1] = wmma_bf16(A1, B1, acc[1][1]);
        acc[1][2] = wmma_bf16(A1, B2, acc[1][2]);
        acc[1][3] = wmma_bf16(A1, B3, acc[1][3]);
    }
}

// =============================================================================
// K0: f32 -> bf16 pack (grid-stride)
// =============================================================================
__global__ void cvt_f32_bf16(const float* __restrict__ src,
                             __bf16* __restrict__ dst, int n) {
    int i = blockIdx.x * blockDim.x + threadIdx.x;
    int stride = gridDim.x * blockDim.x;
    for (; i < n; i += stride) dst[i] = (__bf16)src[i];
}

// =============================================================================
// K1: fused QKV projection.  y = x @ W^T + b   (per batch: 1024x512 @ 512x512)
// grid = (N/128, C/128, 3*B), block = 256 (8 waves, each a 32x64 strip)
// =============================================================================
__global__ void qkv_proj(const __bf16* __restrict__ xb,   // (B,C,N) bf16
                         const __bf16* __restrict__ wb,   // (3,N,N) bf16
                         const float*  __restrict__ bq,
                         const float*  __restrict__ bk,
                         const float*  __restrict__ bv,
                         __bf16* __restrict__ qb,          // (B,C,N) bf16
                         __bf16* __restrict__ kb,          // (B,C,N) bf16
                         float*  __restrict__ vf,          // (B,C,N) f32
                         __bf16* __restrict__ vtb)         // (B,N,C) bf16
{
    const int lane  = threadIdx.x & 31;
    const int wave  = threadIdx.x >> 5;
    const int mBase = blockIdx.y * 128 + (wave & 3) * 32;
    const int nBase = blockIdx.x * 128 + (wave >> 2) * 64;
    const int b     = blockIdx.z % B_;
    const int which = blockIdx.z / B_;

    const __bf16* a0 = xb + ((size_t)b * C_ + mBase + (lane & 15)) * N_;
    const __bf16* a1 = a0 + (size_t)16 * N_;
    // B[k][n] = W[n][k] -> row n of W, contiguous over k
    const __bf16* w0 = wb + ((size_t)which * N_ + nBase + (lane & 15)) * N_;

    v8f acc[2][4] = {};
    wave_gemm_32x64(a0, a1,
                    w0, w0 + (size_t)16 * N_, w0 + (size_t)32 * N_, w0 + (size_t)48 * N_,
                    N_, lane, acc);

    const float* bias = (which == 0) ? bq : (which == 1) ? bk : bv;
    float bia[4];
    #pragma unroll
    for (int j = 0; j < 4; ++j) bia[j] = bias[nBase + j * 16 + (lane & 15)];

    // C/D layout: VGPR r, lane l -> M = r + (l>>4)*8, N = l&15
    const int mrow = mBase + (lane >> 4) * 8;
    if (which < 2) {
        __bf16* dst = (which == 0 ? qb : kb) + (size_t)b * C_ * N_;
        #pragma unroll
        for (int i = 0; i < 2; ++i)
            #pragma unroll
            for (int j = 0; j < 4; ++j)
                for (int r = 0; r < 8; ++r) {
                    const size_t m = (size_t)(mrow + i * 16 + r);
                    dst[m * N_ + nBase + j * 16 + (lane & 15)] =
                        (__bf16)(acc[i][j][r] + bia[j]);
                }
    } else {
        #pragma unroll
        for (int i = 0; i < 2; ++i)
            #pragma unroll
            for (int j = 0; j < 4; ++j)
                for (int r = 0; r < 8; ++r) {
                    const int m = mrow + i * 16 + r;
                    const int n = nBase + j * 16 + (lane & 15);
                    const float f = acc[i][j][r] + bia[j];
                    vf[((size_t)b * C_ + m) * N_ + n]  = f;
                    vtb[((size_t)b * N_ + n) * C_ + m] = (__bf16)f;
                }
    }
}

// =============================================================================
// K2: energy = q @ k^T per batch (1024x512 @ 512x1024), streaming f32 out (NT)
// grid = (C/128, C/128, B), block = 256
// =============================================================================
__global__ void energy_gemm(const __bf16* __restrict__ qb,
                            const __bf16* __restrict__ kb,
                            float* __restrict__ energy)    // (B,C,C) f32
{
    const int lane  = threadIdx.x & 31;
    const int wave  = threadIdx.x >> 5;
    const int mBase = blockIdx.y * 128 + (wave & 3) * 32;   // query row c
    const int nBase = blockIdx.x * 128 + (wave >> 2) * 64;  // key row d
    const int b     = blockIdx.z;

    const __bf16* a0 = qb + ((size_t)b * C_ + mBase + (lane & 15)) * N_;
    const __bf16* a1 = a0 + (size_t)16 * N_;
    // B[k=n][col=d] = k[b, d, n] -> row d of k, contiguous over n
    const __bf16* k0 = kb + ((size_t)b * C_ + nBase + (lane & 15)) * N_;

    v8f acc[2][4] = {};
    wave_gemm_32x64(a0, a1,
                    k0, k0 + (size_t)16 * N_, k0 + (size_t)32 * N_, k0 + (size_t)48 * N_,
                    N_, lane, acc);

    const int mrow = mBase + (lane >> 4) * 8;
    float* erow = energy + (size_t)b * C_ * C_;
    #pragma unroll
    for (int i = 0; i < 2; ++i)
        #pragma unroll
        for (int j = 0; j < 4; ++j)
            for (int r = 0; r < 8; ++r) {
                const size_t m = (size_t)(mrow + i * 16 + r);
                // streaming: written once, consumed once by softmax -> NT hint
                __builtin_nontemporal_store(acc[i][j][r],
                    &erow[m * C_ + nBase + j * 16 + (lane & 15)]);
            }
}

// =============================================================================
// K3: row softmax, in place: read 1024 f32 (NT), write 1024 bf16 probs at the
// same row base (row stride stays C*4 bytes). One wave per row, 8 rows/block.
// =============================================================================
__global__ void softmax_rows(float* __restrict__ energy)
{
    const int lane = threadIdx.x & 31;
    const int wave = threadIdx.x >> 5;
    const size_t row = (size_t)blockIdx.x * 8 + wave;   // < B*C
    float* e = energy + row * C_;

    float vals[32];
    float m = -3.0e38f;
    #pragma unroll
    for (int i = 0; i < 32; ++i) {
        vals[i] = __builtin_nontemporal_load(&e[lane + i * 32]);
        m = fmaxf(m, vals[i]);
    }
    #pragma unroll
    for (int off = 16; off > 0; off >>= 1) m = fmaxf(m, __shfl_xor(m, off, 32));

    float s = 0.0f;
    #pragma unroll
    for (int i = 0; i < 32; ++i) {
        vals[i] = __expf(vals[i] - m);
        s += vals[i];
    }
    #pragma unroll
    for (int off = 16; off > 0; off >>= 1) s += __shfl_xor(s, off, 32);
    const float inv = 1.0f / s;

    __bf16* p = (__bf16*)e;   // probs occupy first 2KB of this row's 4KB
    #pragma unroll
    for (int i = 0; i < 32; ++i) p[lane + i * 32] = (__bf16)(vals[i] * inv);
}

// =============================================================================
// K4: out = beta * (P @ V) + V.   P is bf16 with row stride 2*C elements.
// grid = (N/128, C/128, B), block = 256
// =============================================================================
__global__ void attn_out(const float* __restrict__ energy,  // holds bf16 probs
                         const __bf16* __restrict__ vtb,    // (B,N,C) bf16
                         const float*  __restrict__ vf,     // (B,C,N) f32
                         const float*  __restrict__ beta_p,
                         float* __restrict__ out)           // (B,C,N) f32
{
    const int lane  = threadIdx.x & 31;
    const int wave  = threadIdx.x >> 5;
    const int mBase = blockIdx.y * 128 + (wave & 3) * 32;   // row c
    const int nBase = blockIdx.x * 128 + (wave >> 2) * 64;  // col n
    const int b     = blockIdx.z;
    const float beta = beta_p[0];

    const __bf16* pbase = (const __bf16*)energy;
    const __bf16* a0 = pbase + ((size_t)b * C_ + mBase + (lane & 15)) * (size_t)(2 * C_);
    const __bf16* a1 = a0 + (size_t)16 * (2 * C_);
    // B[d][n] = vT[n][d] -> row n of vtb, contiguous over d
    const __bf16* v0 = vtb + ((size_t)b * N_ + nBase + (lane & 15)) * C_;

    v8f acc[2][4] = {};
    // NOTE: A row stride is 2*C here, but wave_gemm_32x64 only offsets by kk
    // within a row, so the stride is fully carried by a0/a1/v* pointers.
    wave_gemm_32x64(a0, a1,
                    v0, v0 + (size_t)16 * C_, v0 + (size_t)32 * C_, v0 + (size_t)48 * C_,
                    C_, lane, acc);

    const int mrow = mBase + (lane >> 4) * 8;
    #pragma unroll
    for (int i = 0; i < 2; ++i)
        #pragma unroll
        for (int j = 0; j < 4; ++j)
            for (int r = 0; r < 8; ++r) {
                const int m = mrow + i * 16 + r;
                const size_t idx =
                    ((size_t)b * C_ + m) * N_ + nBase + j * 16 + (lane & 15);
                out[idx] = beta * acc[i][j][r] + vf[idx];
            }
}

// =============================================================================
// Launch
// =============================================================================
extern "C" void kernel_launch(void* const* d_in, const int* in_sizes, int n_in,
                              void* d_out, int out_size, void* d_ws, size_t ws_size,
                              hipStream_t stream) {
    const float* x  = (const float*)d_in[0];
    const float* Wq = (const float*)d_in[1];
    const float* bq = (const float*)d_in[2];
    const float* Wk = (const float*)d_in[3];
    const float* bk = (const float*)d_in[4];
    const float* Wv = (const float*)d_in[5];
    const float* bv = (const float*)d_in[6];
    const float* beta = (const float*)d_in[7];
    float* out = (float*)d_out;

    // ---- workspace layout (bytes) ----
    char* ws = (char*)d_ws;
    size_t off = 0;
    __bf16* xb  = (__bf16*)(ws + off); off += (size_t)B_ * C_ * N_ * 2;  // 32 MB
    __bf16* wb  = (__bf16*)(ws + off); off += (size_t)3 * N_ * N_ * 2;   // 1.5 MB
    __bf16* qb  = (__bf16*)(ws + off); off += (size_t)B_ * C_ * N_ * 2;  // 32 MB
    __bf16* kb  = (__bf16*)(ws + off); off += (size_t)B_ * C_ * N_ * 2;  // 32 MB
    __bf16* vtb = (__bf16*)(ws + off); off += (size_t)B_ * C_ * N_ * 2;  // 32 MB
    float*  vf  = (float*) (ws + off); off += (size_t)B_ * C_ * N_ * 4;  // 64 MB
    float*  en  = (float*) (ws + off); off += (size_t)B_ * C_ * C_ * 4;  // 128 MB

    // K0: pack to bf16
    const int nx = B_ * C_ * N_;
    const int nw = N_ * N_;
    cvt_f32_bf16<<<8192, 256, 0, stream>>>(x,  xb,          nx);
    cvt_f32_bf16<<<1024, 256, 0, stream>>>(Wq, wb,          nw);
    cvt_f32_bf16<<<1024, 256, 0, stream>>>(Wk, wb + nw,     nw);
    cvt_f32_bf16<<<1024, 256, 0, stream>>>(Wv, wb + 2 * nw, nw);

    // K1: q/k/v projections (128x128 tile per workgroup)
    qkv_proj<<<dim3(N_ / 128, C_ / 128, 3 * B_), 256, 0, stream>>>(
        xb, wb, bq, bk, bv, qb, kb, vf, vtb);

    // K2: energy = q k^T
    energy_gemm<<<dim3(C_ / 128, C_ / 128, B_), 256, 0, stream>>>(qb, kb, en);

    // K3: softmax (in place, f32 -> bf16 probs)
    softmax_rows<<<(B_ * C_) / 8, 256, 0, stream>>>(en);

    // K4: out = beta * (P V) + V
    attn_out<<<dim3(N_ / 128, C_ / 128, B_), 256, 0, stream>>>(en, vtb, vf, beta, out);
}
